// BiRNN_46282567582014
// MI455X (gfx1250) — matrix operand order repaired and checked by reading, not compiled, not run on hardware
//
#include <hip/hip_runtime.h>
#include <hip/hip_bf16.h>

typedef unsigned int u32;
typedef __attribute__((ext_vector_type(16))) __bf16 v16bf;
typedef __attribute__((ext_vector_type(8)))  float  v8f;

#define Bz 32
#define Tz 2048
#define Dz 512
#define Hz 256
#define G4 1024   // 4*H

// ---------- helpers ----------
__device__ __forceinline__ unsigned short f2bf(float f) {
    u32 u = __float_as_uint(f);
    u32 r = u + 0x7FFFu + ((u >> 16) & 1u);   // round-to-nearest-even
    return (unsigned short)(r >> 16);
}

// A-matrix (16x32 bf16) per documented CDNA5 layout.
// row = per-lane pointer to this lane's M row (M = lane&15), kbase = ks*32 + (lane&16 ? 8 : 0)
__device__ __forceinline__ v16bf load_tileA(const unsigned short* row, int kbase) {
    union { u32 u[8]; v16bf v; } U;
#pragma unroll
    for (int q = 0; q < 8; ++q) {
        int k = kbase + ((q & 3) << 1) + ((q >> 2) << 4);
        U.u[q] = *(const u32*)(row + k);
    }
    return U.v;
}

// B-matrix (32x16 bf16): row = pointer to weight row n (N = lane&15), kbase = ks*32 + (lane&16 ? 8 : 0)
__device__ __forceinline__ v16bf load_tileB(const unsigned short* row, int kbase) {
    union { u32 u[8]; v16bf v; } U;
#pragma unroll
    for (int q = 0; q < 8; ++q) U.u[q] = *(const u32*)(row + kbase + ((q & 3) << 1) + ((q >> 2) << 4));
    return U.v;
}

__device__ __forceinline__ v8f wmma_bf16(v16bf a, v16bf b, v8f c) {
    return __builtin_amdgcn_wmma_f32_16x16x32_bf16(false, a, false, b, (short)0, c, false, false);
}

__device__ __forceinline__ float sigm(float x) { return 1.0f / (1.0f + __expf(-x)); }

// ---------- kernel 1: ragged lengths ----------
__global__ __launch_bounds__(256) void k_lengths(const float* __restrict__ x, int* __restrict__ lengths) {
    int b = blockIdx.x, tid = threadIdx.x;
    __shared__ int cnt;
    if (tid == 0) cnt = 0;
    __syncthreads();
    int local = 0;
    for (int t = tid; t < Tz; t += 256) {
        const float* row = x + ((size_t)b * Tz + t) * Dz;
        bool nz = false;
        for (int k = 0; k < Dz && !nz; k += 4) {
            float4 v = *(const float4*)(row + k);
            nz = (v.x != 0.f) | (v.y != 0.f) | (v.z != 0.f) | (v.w != 0.f);
        }
        local += nz ? 1 : 0;
    }
    atomicAdd(&cnt, local);
    __syncthreads();
    if (tid == 0) lengths[b] = cnt;
}

// ---------- kernel 2: f32 -> bf16 ----------
__global__ __launch_bounds__(256) void k_cvt_bf16(const float* __restrict__ src,
                                                  unsigned short* __restrict__ dst, long n) {
    long i = (long)blockIdx.x * blockDim.x + threadIdx.x;
    long stride = (long)gridDim.x * blockDim.x;
    for (; i < n; i += stride) dst[i] = f2bf(src[i]);
}

// ---------- kernel 3: input projection GEMM (both directions) ----------
// grid (BT/32, G4/64, 2), block 128 (4 waves: wm = w&1 -> M tile, wn = w>>1 -> 32-col strip)
__global__ __launch_bounds__(128) void k_input_gemm(
    const unsigned short* __restrict__ xbf,
    const unsigned short* __restrict__ Wihf, const unsigned short* __restrict__ Wihb,
    const float* __restrict__ bihf, const float* __restrict__ bhhf,
    const float* __restrict__ bihb, const float* __restrict__ bhhb,
    const int* __restrict__ lengths,
    float* __restrict__ Zf, float* __restrict__ Zb) {

    int mblk = blockIdx.x, nblk = blockIdx.y, dir = blockIdx.z;
    int tid = threadIdx.x, w = tid >> 5, lane = tid & 31;
    int wm = w & 1, wn = w >> 1;

    const unsigned short* Wih = dir ? Wihb : Wihf;
    const float* bih = dir ? bihb : bihf;
    const float* bhh = dir ? bhhb : bhhf;
    float* Z = dir ? Zb : Zf;

    // A row for this lane (flat row r = b*T + t); backward gathers the flipped row
    int m = lane & 15;
    int r = mblk * 32 + wm * 16 + m;
    int b = r >> 11, t = r & (Tz - 1);
    long asrc = (long)r * Dz;
    if (dir) {
        int len = lengths[b];
        if (t < len) asrc = ((long)b * Tz + (len - 1 - t)) * Dz;  // else row is zero in x anyway
    }
    const unsigned short* arow = xbf + asrc;
    int kbA = (lane & 16) ? 8 : 0;
    int kbB = (lane & 16) ? 8 : 0;

    // B rows + biases for the two 16-col sub-tiles of this wave
    int n0 = nblk * 64 + wn * 32 + (lane & 15);
    int n1 = n0 + 16;
    const unsigned short* brow0 = Wih + (size_t)n0 * Dz;
    const unsigned short* brow1 = Wih + (size_t)n1 * Dz;
    float bias0 = bih[n0] + bhh[n0];
    float bias1 = bih[n1] + bhh[n1];

    v8f acc0 = {}, acc1 = {};
#pragma unroll
    for (int ks = 0; ks < Dz / 32; ++ks) {
        v16bf A  = load_tileA(arow,  ks * 32 + kbA);
        v16bf B0 = load_tileB(brow0, ks * 32 + kbB);
        v16bf B1 = load_tileB(brow1, ks * 32 + kbB);
        acc0 = wmma_bf16(A, B0, acc0);
        acc1 = wmma_bf16(A, B1, acc1);
    }
#pragma unroll
    for (int q = 0; q < 8; ++q) {
        int row = mblk * 32 + wm * 16 + q + ((lane & 16) ? 8 : 0);
        Z[(size_t)row * G4 + n0] = acc0[q] + bias0;
        Z[(size_t)row * G4 + n1] = acc1[q] + bias1;
    }
}

// ---------- grid sync (monotonic counter) ----------
__device__ __forceinline__ void gsync(int* ctr, int nwg, int phase, int tid) {
    __threadfence();
    __syncthreads();
    if (tid == 0) {
        __hip_atomic_fetch_add(ctr, 1, __ATOMIC_ACQ_REL, __HIP_MEMORY_SCOPE_AGENT);
        while (__hip_atomic_load(ctr, __ATOMIC_ACQUIRE, __HIP_MEMORY_SCOPE_AGENT) < nwg * phase)
            __builtin_amdgcn_s_sleep(2);
    }
    __syncthreads();
}

// ---------- kernel 4: persistent recurrence ----------
// grid 32 (dir = blk>>4, g = blk&15), block 256 (8 waves).
// WG g owns 64 gate columns = gates {i,f,g,o} x hidden j in [g*16, g*16+16).
// Whh slice resident in LDS; h double-buffered in global (bf16, [32][256] row-major per slot).
#define HSLOT (Bz * Hz)
__global__ __launch_bounds__(256) void k_recurrent(
    const float* __restrict__ Zf, const float* __restrict__ Zb,
    const unsigned short* __restrict__ Whhf, const unsigned short* __restrict__ Whhb,
    unsigned short* __restrict__ hf, unsigned short* __restrict__ hb,
    int* __restrict__ counters, float* __restrict__ out) {

    int dir = blockIdx.x >> 4, g = blockIdx.x & 15;
    int tid = threadIdx.x, w = tid >> 5, lane = tid & 31;

    const float* Z = dir ? Zb : Zf;
    const unsigned short* Whh = dir ? Whhb : Whhf;
    unsigned short* hbuf = dir ? hb : hf;
    int* ctr = counters + dir * 16;   // separate cachelines

    __shared__ unsigned short Wl[64 * Hz];  // 32 KB: [c][k], c = gate*16 + jj
    __shared__ float zl[Bz * 64];           // 8 KB gate pre-activations
    __shared__ float cl[Bz * 16];           // 2 KB cell state

    // load Whh slice into LDS (u32 granules)
    for (int i = tid; i < 64 * Hz / 2; i += 256) {
        int c = i >> 7, kw = i & 127;                    // 128 u32 per 256-elem row
        int n = (c >> 4) * Hz + g * 16 + (c & 15);       // global gate row
        ((u32*)Wl)[i] = *(const u32*)(Whh + (size_t)n * Hz + kw * 2);
    }
    // zero cell state and h slot 0 (our slice)
    for (int i = tid; i < Bz * 16; i += 256) cl[i] = 0.f;
    for (int i = tid; i < Bz * 16; i += 256) {
        int b = i >> 4, jj = i & 15;
        hbuf[b * Hz + g * 16 + jj] = 0;
    }
    gsync(ctr, 16, 1, tid);

    // static per-wave tile assignment: 2 M-tiles x 4 N-tiles (N-tile == gate)
    int mt = w & 1, nt = w >> 1;
    int kbA = (lane & 16) ? 8 : 0;
    int kbB = (lane & 16) ? 8 : 0;
    const unsigned short* brow = Wl + (size_t)(nt * 16 + (lane & 15)) * Hz;
    int nglob = nt * Hz + g * 16 + (lane & 15);
    int arow_off = (mt * 16 + (lane & 15)) * Hz;

    for (int t = 0; t < Tz; ++t) {
        const unsigned short* hread = hbuf + ((t & 1) ? HSLOT : 0) + arow_off;
        unsigned short* hwrite = hbuf + ((t & 1) ? 0 : HSLOT);

        // init accum from precomputed input projection (bias already folded in)
        v8f acc;
#pragma unroll
        for (int q = 0; q < 8; ++q) {
            int b = mt * 16 + q + ((lane & 16) ? 8 : 0);
            acc[q] = Z[((size_t)b * Tz + t) * G4 + nglob];
        }
        // z += h @ Whh^T (slice)
#pragma unroll
        for (int ks = 0; ks < Hz / 32; ++ks) {
            v16bf A = load_tileA(hread, ks * 32 + kbA);
            v16bf B = load_tileB(brow,  ks * 32 + kbB);
            acc = wmma_bf16(A, B, acc);
        }
#pragma unroll
        for (int q = 0; q < 8; ++q) {
            int b = mt * 16 + q + ((lane & 16) ? 8 : 0);
            zl[b * 64 + nt * 16 + (lane & 15)] = acc[q];
        }
        __syncthreads();

        // gate combine + state update: 512 elems, 2 per thread
#pragma unroll
        for (int e = 0; e < 2; ++e) {
            int idx = tid + e * 256;
            int b = idx >> 4, jj = idx & 15;
            float zi = zl[b * 64 +  0 + jj];
            float zfv = zl[b * 64 + 16 + jj];
            float zg = zl[b * 64 + 32 + jj];
            float zo = zl[b * 64 + 48 + jj];
            float c0 = cl[idx];
            float cn = sigm(zfv) * c0 + sigm(zi) * tanhf(zg);
            float hh = sigm(zo) * tanhf(cn);
            cl[idx] = cn;
            int j = g * 16 + jj;
            hwrite[b * Hz + j] = f2bf(hh);
            out[((size_t)b * Tz + t) * (2 * Hz) + dir * Hz + j] = hh;
        }
        gsync(ctr, 16, t + 2, tid);   // publish h(t+1) to all WGs of this direction
    }
}

// ---------- host ----------
extern "C" void kernel_launch(void* const* d_in, const int* in_sizes, int n_in,
                              void* d_out, int out_size, void* d_ws, size_t ws_size,
                              hipStream_t stream) {
    const float* x     = (const float*)d_in[0];
    const float* Wf_ih = (const float*)d_in[1];
    const float* Wf_hh = (const float*)d_in[2];
    const float* bf_ih = (const float*)d_in[3];
    const float* bf_hh = (const float*)d_in[4];
    const float* Wb_ih = (const float*)d_in[5];
    const float* Wb_hh = (const float*)d_in[6];
    const float* bb_ih = (const float*)d_in[7];
    const float* bb_hh = (const float*)d_in[8];

    char* ws = (char*)d_ws;
    size_t off = 0;
    auto take = [&](size_t bytes) { size_t o = off; off = (off + bytes + 255) & ~(size_t)255; return o; };

    size_t o_ctr   = take(256);
    size_t o_len   = take(Bz * sizeof(int));
    size_t o_hf    = take(2 * HSLOT * sizeof(unsigned short));
    size_t o_hb    = take(2 * HSLOT * sizeof(unsigned short));
    size_t o_wfih  = take((size_t)G4 * Dz * 2);
    size_t o_wbih  = take((size_t)G4 * Dz * 2);
    size_t o_wfhh  = take((size_t)G4 * Hz * 2);
    size_t o_wbhh  = take((size_t)G4 * Hz * 2);
    size_t o_xbf   = take((size_t)Bz * Tz * Dz * 2);
    size_t o_zf    = take((size_t)Bz * Tz * G4 * 4);
    size_t o_zb    = take((size_t)Bz * Tz * G4 * 4);
    if (off > ws_size) return;  // workspace too small; nothing safe to do

    int*            ctr   = (int*)(ws + o_ctr);
    int*            len   = (int*)(ws + o_len);
    unsigned short* hf    = (unsigned short*)(ws + o_hf);
    unsigned short* hb    = (unsigned short*)(ws + o_hb);
    unsigned short* wfih  = (unsigned short*)(ws + o_wfih);
    unsigned short* wbih  = (unsigned short*)(ws + o_wbih);
    unsigned short* wfhh  = (unsigned short*)(ws + o_wfhh);
    unsigned short* wbhh  = (unsigned short*)(ws + o_wbhh);
    unsigned short* xbf   = (unsigned short*)(ws + o_xbf);
    float*          Zf    = (float*)(ws + o_zf);
    float*          Zb    = (float*)(ws + o_zb);
    float*          out   = (float*)d_out;

    hipMemsetAsync(ws + o_ctr, 0, 256, stream);

    k_lengths<<<Bz, 256, 0, stream>>>(x, len);

    k_cvt_bf16<<<4096, 256, 0, stream>>>(x,     xbf,  (long)Bz * Tz * Dz);
    k_cvt_bf16<<<512,  256, 0, stream>>>(Wf_ih, wfih, (long)G4 * Dz);
    k_cvt_bf16<<<512,  256, 0, stream>>>(Wb_ih, wbih, (long)G4 * Dz);
    k_cvt_bf16<<<256,  256, 0, stream>>>(Wf_hh, wfhh, (long)G4 * Hz);
    k_cvt_bf16<<<256,  256, 0, stream>>>(Wb_hh, wbhh, (long)G4 * Hz);

    dim3 ggrid(Bz * Tz / 32, G4 / 64, 2);
    k_input_gemm<<<ggrid, 128, 0, stream>>>(xbf, wfih, wbih, bf_ih, bf_hh, bb_ih, bb_hh,
                                            len, Zf, Zb);

    k_recurrent<<<32, 256, 0, stream>>>(Zf, Zb, wfhh, wbhh, hf, hb, ctr, out);
}
